// QuantLinear_3521873183240
// MI455X (gfx1250) — compile-verified, matrix operand order checked by reading, not run on hardware
//
#include <hip/hip_runtime.h>
#include <stdint.h>

// ---------------------------------------------------------------------------
// GPTQ 4-bit quantized GEMM for gfx1250 (MI455X), wave32 + WMMA + async-to-LDS.
//   out[M,N] = x[M,K] * dequant(qweight,scales,qzeros)[K,N] + bias
//   M=4096, K=4096, N=11008, GROUPSIZE=128, 8x 4-bit per int32.
// Path A (preferred, needs ws >= M*K*2 bytes):
//   1) convert_x_f16: x f32 -> f16 into workspace (memory-bound pre-pass)
//   2) GEMM: A tiles via GLOBAL_LOAD_ASYNC_TO_LDS_B128 (ASYNCcnt), B tiles
//      dequantized in VALU with f16 magic-number trick, 16 wmma/wave/k-step.
// Path B (fallback): register-staged A with f32->f16 conversion in-kernel.
// Block tile: 256x128x32, 256 threads (8 waves), wave tile 64x64 (4x4 wmma),
// double-buffered LDS, one barrier per k-step.
// ---------------------------------------------------------------------------

typedef _Float16 half16 __attribute__((ext_vector_type(16)));
typedef _Float16 half8  __attribute__((ext_vector_type(8)));
typedef _Float16 half2t __attribute__((ext_vector_type(2)));
typedef float    f32x4  __attribute__((ext_vector_type(4)));
typedef float    f32x8  __attribute__((ext_vector_type(8)));

#define MAXQ      15
#define PACK      8
#define GROUPSIZE 128
#define M_DIM     4096
#define K_DIM     4096
#define N_DIM     11008

#define BM 256
#define BN 128
#define BK 32
#define LDS_STRIDE 40            // halves: 80B = 20 banks, multiple of 16B
#define A_BUF (BM * LDS_STRIDE)  // halves per A buffer
#define B_BUF (BN * LDS_STRIDE)  // halves per B buffer

union FragH {
  half16 v;
  half8  h[2];
};

union Half8Pack {
  half8  v8;
  half2t v2[4];
};

union U32H2 {
  unsigned u;
  half2t   h;
};

struct BStage {
  int    qw[2];
  half2t sp[2];   // splat(s / 64)
  half2t op[2];   // splat(-s * (z + 17))
};

// ---------------------------------------------------------------------------
// CDNA5 async global->LDS copy (GLOBAL_LOAD_ASYNC_TO_LDS_B128, ASYNCcnt).
// lds_byte_off is the byte offset within the block's LDS allocation.
// ---------------------------------------------------------------------------
__device__ __forceinline__ void async_ld_b128(unsigned lds_byte_off, uint64_t gaddr) {
  asm volatile("global_load_async_to_lds_b128 %0, %1, off"
               :: "v"(lds_byte_off), "v"(gaddr) : "memory");
}
__device__ __forceinline__ void wait_async0() {
  asm volatile("s_wait_asynccnt 0x0" ::: "memory");
}

// ---------------------------------------------------------------------------
// B tile: 32(K) x 128(N) 4-bit values; quant group constant over the k-tile.
// Magic-number dequant: m = 0x6400 | (w << 6) is f16(1024 + 64w), so
//   (1024 + 64w) * (s/64) + (-s*(z+17)) = s*(w - z - 1)   (z stored - 1)
// One v_pk_fma_f16 per element pair.
// ---------------------------------------------------------------------------
__device__ __forceinline__ void stage_b_load(const int* __restrict__ qweight,
                                             const float* __restrict__ scales,
                                             const int* __restrict__ qzeros,
                                             int k0, int n0, int tid, BStage& st) {
  const int g = k0 / GROUPSIZE;
  #pragma unroll
  for (int t = 0; t < 2; ++t) {
    const int idx  = tid + t * 256;    // 0..511
    const int kp   = idx >> 7;         // packed qweight row within tile (8 k each)
    const int nn   = idx & 127;
    const int ncol = n0 + nn;
    st.qw[t] = qweight[(size_t)(k0 / PACK + kp) * N_DIM + ncol];
    const float s = scales[(size_t)g * N_DIM + ncol];
    const int  zq = (qzeros[(size_t)g * (N_DIM / PACK) + (ncol >> 3)]
                     >> ((ncol & 7) * 4)) & MAXQ;
    const _Float16 sh = (_Float16)(s * 0.015625f);          // s / 64
    const _Float16 oh = (_Float16)(-s * (float)(zq + 17));  // -s*(z+17)
    st.sp[t][0] = sh; st.sp[t][1] = sh;
    st.op[t][0] = oh; st.op[t][1] = oh;
  }
}

__device__ __forceinline__ void stage_b_store(_Float16* __restrict__ ldsb,
                                              int tid, const BStage& st) {
  #pragma unroll
  for (int t = 0; t < 2; ++t) {
    const int idx = tid + t * 256;
    const int kp  = idx >> 7;
    const int nn  = idx & 127;
    Half8Pack hp;
    #pragma unroll
    for (int p = 0; p < 4; ++p) {
      const unsigned tt = (unsigned)st.qw[t] >> (8 * p);
      U32H2 m;
      m.u = ((tt & 0x0Fu) << 6) | ((tt & 0xF0u) << 18) | 0x64006400u;
      hp.v2[p] = m.h * st.sp[t] + st.op[t];   // v_pk_fma_f16
    }
    *(half8*)(ldsb + nn * LDS_STRIDE + kp * 8) = hp.v8;
  }
}

// ---------------------------------------------------------------------------
// A tile (async path): 256 rows x 32 halves from pre-converted f16 x.
// 1024 16B chunks, 4 async b128 copies per thread, padded LDS layout.
// ---------------------------------------------------------------------------
__device__ __forceinline__ void stage_a_async(const _Float16* __restrict__ xh,
                                              int m0, int k0, int tid,
                                              unsigned ldsa_byte_base) {
  #pragma unroll
  for (int t = 0; t < 4; ++t) {
    const int idx = tid + t * 256;     // 0..1023
    const int row = idx >> 2;          // 0..255
    const int c   = idx & 3;           // 16B chunk within row
    const uint64_t ga = (uint64_t)(xh + (size_t)(m0 + row) * K_DIM + k0 + c * 8);
    const unsigned la = ldsa_byte_base + (unsigned)(row * LDS_STRIDE + c * 8) * 2u;
    async_ld_b128(la, ga);
  }
}

// ---------------------------------------------------------------------------
// A tile (fallback path): register-staged f32 -> f16
// ---------------------------------------------------------------------------
__device__ __forceinline__ void stage_a_load(const float* __restrict__ x,
                                             int m0, int k0, int tid, f32x4 r[8]) {
  const int row  = tid >> 1;
  const int colc = (tid & 1) * 16;
  const float* s0 = x + (size_t)(m0 + row) * K_DIM + (k0 + colc);
  const float* s1 = x + (size_t)(m0 + row + 128) * K_DIM + (k0 + colc);
  #pragma unroll
  for (int q = 0; q < 4; ++q) {
    r[q]     = *(const f32x4*)(s0 + 4 * q);
    r[4 + q] = *(const f32x4*)(s1 + 4 * q);
  }
}

__device__ __forceinline__ void stage_a_store(_Float16* __restrict__ ldsa,
                                              int tid, const f32x4 r[8]) {
  const int row  = tid >> 1;
  const int colc = (tid & 1) * 16;
  #pragma unroll
  for (int h = 0; h < 2; ++h) {
    half8 h0, h1;
    #pragma unroll
    for (int t = 0; t < 4; ++t) {
      h0[t]     = (_Float16)r[4 * h + 0][t];
      h0[4 + t] = (_Float16)r[4 * h + 1][t];
      h1[t]     = (_Float16)r[4 * h + 2][t];
      h1[4 + t] = (_Float16)r[4 * h + 3][t];
    }
    _Float16* d = ldsa + (row + h * 128) * LDS_STRIDE + colc;
    *(half8*)(d)     = h0;
    *(half8*)(d + 8) = h1;
  }
}

// ---------------------------------------------------------------------------
// x f32 -> f16 pre-pass (memory-bound: reads 67MB, writes 33MB once)
// ---------------------------------------------------------------------------
__global__ __launch_bounds__(256) void convert_x_f16(const float* __restrict__ x,
                                                     _Float16* __restrict__ xh) {
  const size_t i = ((size_t)blockIdx.x * 256 + threadIdx.x) * 8;
  const f32x4 a = *(const f32x4*)(x + i);
  const f32x4 b = *(const f32x4*)(x + i + 4);
  half8 h;
  #pragma unroll
  for (int t = 0; t < 4; ++t) {
    h[t]     = (_Float16)a[t];
    h[4 + t] = (_Float16)b[t];
  }
  *(half8*)(xh + i) = h;
}

// ---------------------------------------------------------------------------
// Main GEMM (async A path)
// ---------------------------------------------------------------------------
__global__ __launch_bounds__(256) void gptq_gemm_wmma_async(
    const _Float16* __restrict__ xh,
    const int*      __restrict__ qweight,
    const float*    __restrict__ scales,
    const int*      __restrict__ qzeros,
    const float*    __restrict__ bias,
    float*          __restrict__ out)
{
  __shared__ _Float16 smem[2 * A_BUF + 2 * B_BUF];

  const int tid  = threadIdx.x;
  const int lane = tid & 31;
  const int wave = tid >> 5;
  const int hlf  = lane >> 4;
  const int l16  = lane & 15;

  const int n0 = blockIdx.x * BN;
  const int m0 = blockIdx.y * BM;

  const int wm_base = (wave >> 1) * 64;
  const int wn_base = (wave & 1) * 64;

  f32x8 acc[4][4];
  #pragma unroll
  for (int i = 0; i < 4; ++i)
    #pragma unroll
    for (int j = 0; j < 4; ++j)
      #pragma unroll
      for (int e = 0; e < 8; ++e)
        acc[i][j][e] = 0.0f;

  // ---- prologue: stage tile k0=0 into buffer 0
  BStage bst;
  stage_a_async(xh, m0, 0, tid, /*ldsa_byte_base=*/0u);
  stage_b_load(qweight, scales, qzeros, 0, n0, tid, bst);
  stage_b_store(smem + 2 * A_BUF, tid, bst);
  wait_async0();
  __syncthreads();

  int p = 0;
  for (int k0 = 0; k0 < K_DIM; k0 += BK) {
    const bool has_next = (k0 + BK) < K_DIM;

    // issue next A tile's async copies into the alternate buffer (safe:
    // that buffer's readers finished before the barrier that ended last iter)
    if (has_next) {
      stage_a_async(xh, m0, k0 + BK, tid, (unsigned)((p ^ 1) * A_BUF * 2));
      stage_b_load(qweight, scales, qzeros, k0 + BK, n0, tid, bst);
      if (k0 + 2 * BK < K_DIM) {   // L2 prefetch two tiles ahead
        __builtin_prefetch(qweight + (size_t)((k0 + 2 * BK) / PACK) * N_DIM + n0 + tid, 0, 1);
      }
    }

    // ---- fragments from current buffer
    const _Float16* pa = smem + p * A_BUF;
    const _Float16* pb = smem + 2 * A_BUF + p * B_BUF;
    FragH afrag[4], bfrag[4];
    #pragma unroll
    for (int i = 0; i < 4; ++i) {
      const _Float16* base = pa + (wm_base + i * 16 + l16) * LDS_STRIDE;
      afrag[i].h[0] = *(const half8*)(base + hlf * 8);
      afrag[i].h[1] = *(const half8*)(base + 16 + hlf * 8);
    }
    #pragma unroll
    for (int j = 0; j < 4; ++j) {
      const _Float16* base = pb + (wn_base + j * 16 + l16) * LDS_STRIDE;
      bfrag[j].h[0] = *(const half8*)(base + hlf * 8);
      bfrag[j].h[1] = *(const half8*)(base + 16 + hlf * 8);
    }

    // ---- 16x v_wmma_f32_16x16x32_f16 per wave per k-step
    #pragma unroll
    for (int i = 0; i < 4; ++i)
      #pragma unroll
      for (int j = 0; j < 4; ++j)
        acc[i][j] = __builtin_amdgcn_wmma_f32_16x16x32_f16(
            false, afrag[i].v, false, bfrag[j].v,
            (short)0, acc[i][j], false, false);

    if (has_next) {
      stage_b_store(smem + 2 * A_BUF + (p ^ 1) * B_BUF, tid, bst);
      wait_async0();        // our async A copies landed
      __syncthreads();      // everyone's copies + B stores visible
      p ^= 1;
    }
  }

  // ---- epilogue: C/D layout -> M = v + 8*hlf, N = l16; add bias
  #pragma unroll
  for (int i = 0; i < 4; ++i) {
    #pragma unroll
    for (int j = 0; j < 4; ++j) {
      const int col = n0 + wn_base + j * 16 + l16;
      const float b = bias[col];
      #pragma unroll
      for (int v = 0; v < 8; ++v) {
        const int row = m0 + wm_base + i * 16 + v + hlf * 8;
        out[(size_t)row * N_DIM + col] = acc[i][j][v] + b;
      }
    }
  }
}

// ---------------------------------------------------------------------------
// Fallback GEMM (no workspace): register-staged A conversion
// ---------------------------------------------------------------------------
__global__ __launch_bounds__(256) void gptq_gemm_wmma_direct(
    const float* __restrict__ x,
    const int*   __restrict__ qweight,
    const float* __restrict__ scales,
    const int*   __restrict__ qzeros,
    const float* __restrict__ bias,
    float*       __restrict__ out)
{
  __shared__ _Float16 lds_a[2 * A_BUF];
  __shared__ _Float16 lds_b[2 * B_BUF];

  const int tid  = threadIdx.x;
  const int lane = tid & 31;
  const int wave = tid >> 5;
  const int hlf  = lane >> 4;
  const int l16  = lane & 15;

  const int n0 = blockIdx.x * BN;
  const int m0 = blockIdx.y * BM;
  const int wm_base = (wave >> 1) * 64;
  const int wn_base = (wave & 1) * 64;

  f32x8 acc[4][4];
  #pragma unroll
  for (int i = 0; i < 4; ++i)
    #pragma unroll
    for (int j = 0; j < 4; ++j)
      #pragma unroll
      for (int e = 0; e < 8; ++e)
        acc[i][j][e] = 0.0f;

  f32x4  areg[8];
  BStage bst;
  stage_a_load(x, m0, 0, tid, areg);
  stage_b_load(qweight, scales, qzeros, 0, n0, tid, bst);
  stage_a_store(lds_a, tid, areg);
  stage_b_store(lds_b, tid, bst);
  __syncthreads();

  int p = 0;
  for (int k0 = 0; k0 < K_DIM; k0 += BK) {
    const bool has_next = (k0 + BK) < K_DIM;
    if (has_next) {
      stage_a_load(x, m0, k0 + BK, tid, areg);
      stage_b_load(qweight, scales, qzeros, k0 + BK, n0, tid, bst);
    }

    const _Float16* pa = lds_a + p * A_BUF;
    const _Float16* pb = lds_b + p * B_BUF;
    FragH afrag[4], bfrag[4];
    #pragma unroll
    for (int i = 0; i < 4; ++i) {
      const _Float16* base = pa + (wm_base + i * 16 + l16) * LDS_STRIDE;
      afrag[i].h[0] = *(const half8*)(base + hlf * 8);
      afrag[i].h[1] = *(const half8*)(base + 16 + hlf * 8);
    }
    #pragma unroll
    for (int j = 0; j < 4; ++j) {
      const _Float16* base = pb + (wn_base + j * 16 + l16) * LDS_STRIDE;
      bfrag[j].h[0] = *(const half8*)(base + hlf * 8);
      bfrag[j].h[1] = *(const half8*)(base + 16 + hlf * 8);
    }

    #pragma unroll
    for (int i = 0; i < 4; ++i)
      #pragma unroll
      for (int j = 0; j < 4; ++j)
        acc[i][j] = __builtin_amdgcn_wmma_f32_16x16x32_f16(
            false, afrag[i].v, false, bfrag[j].v,
            (short)0, acc[i][j], false, false);

    if (has_next) {
      stage_a_store(lds_a + (p ^ 1) * A_BUF, tid, areg);
      stage_b_store(lds_b + (p ^ 1) * B_BUF, tid, bst);
      __syncthreads();
      p ^= 1;
    }
  }

  #pragma unroll
  for (int i = 0; i < 4; ++i) {
    #pragma unroll
    for (int j = 0; j < 4; ++j) {
      const int col = n0 + wn_base + j * 16 + l16;
      const float b = bias[col];
      #pragma unroll
      for (int v = 0; v < 8; ++v) {
        const int row = m0 + wm_base + i * 16 + v + hlf * 8;
        out[(size_t)row * N_DIM + col] = acc[i][j][v] + b;
      }
    }
  }
}

extern "C" void kernel_launch(void* const* d_in, const int* in_sizes, int n_in,
                              void* d_out, int out_size, void* d_ws, size_t ws_size,
                              hipStream_t stream) {
  // setup_inputs() order: x, qweight, scales, qzeros, g_idx, bias
  const float* x       = (const float*)d_in[0];
  const int*   qweight = (const int*)  d_in[1];
  const float* scales  = (const float*)d_in[2];
  const int*   qzeros  = (const int*)  d_in[3];
  // d_in[4] = g_idx: contiguous groups (k/128), folded into indexing.
  const float* bias    = (const float*)d_in[5];
  float*       out     = (float*)d_out;

  const size_t xh_bytes = (size_t)M_DIM * K_DIM * sizeof(_Float16);
  dim3 grid(N_DIM / BN, M_DIM / BM);   // (86, 16)

  if (ws_size >= xh_bytes) {
    _Float16* xh = (_Float16*)d_ws;
    convert_x_f16<<<dim3((M_DIM * (size_t)K_DIM) / (256 * 8)), dim3(256), 0, stream>>>(x, xh);
    gptq_gemm_wmma_async<<<grid, dim3(256), 0, stream>>>(xh, qweight, scales, qzeros, bias, out);
  } else {
    gptq_gemm_wmma_direct<<<grid, dim3(256), 0, stream>>>(x, qweight, scales, qzeros, bias, out);
  }
}
